// Discriminator_5669356833707
// MI455X (gfx1250) — matrix operand order repaired and checked by reading, not compiled, not run on hardware
//
#include <hip/hip_runtime.h>

// MolGAN discriminator forward for MI455X (gfx1250, wave32, WMMA).
// One workgroup (256 threads = 8 wave32s) per batch element.
// All GEMMs: v_wmma_f32_16x16x32_bf16 (f32 accumulate).
// All B-side matrices stored TRANSPOSED (N x K) in LDS so that every
// fragment load / transposed D-store is a contiguous 16B ds_load/store_b128.
// bf16 conversions use native casts (hardware cvt on gfx1250).

typedef __attribute__((ext_vector_type(16))) __bf16 v16bf;
typedef __attribute__((ext_vector_type(8)))  float  v8f;

#define NN    128   // nodes
#define STR   136   // activation LDS row stride (bf16): 272B rows, conflict-free, 16B aligned
#define WSTR  96    // pooled-weight LDS K stride (K padded 69 -> 96)

union frag16 { uint4 q[2]; v16bf f; __bf16 h[16]; };
union pack8  { uint4 q;    __bf16 h[8]; };
union pack4  { uint2 q;    __bf16 h[4]; };

__device__ __forceinline__ __bf16 f2bf16(float f) { return (__bf16)f; }
__device__ __forceinline__ float  bf2f(__bf16 b)  { return (float)b; }

__device__ __forceinline__ v8f wmma_bf16(v16bf a, v16bf b, v8f c) {
    return __builtin_amdgcn_wmma_f32_16x16x32_bf16(
        false, a, false, b, (short)0, c, false, false);
}

// A fragment (16x32 MxK) from row-major (M x K, ld) LDS tile at `base`.
// lane: m = lane&15, kb = (lane>=16)?8:0; elems 0..7 = K kb..kb+7, 8..15 = K 16+kb..23+kb.
__device__ __forceinline__ v16bf load_a_rm(const __bf16* base, int ld) {
    const int lane = threadIdx.x & 31;
    const __bf16* p = base + (lane & 15) * ld + ((lane >> 4) << 3);
    frag16 u;
    u.q[0] = *(const uint4*)(p);
    u.q[1] = *(const uint4*)(p + 16);
    return u.f;
}

// B fragment (32x16 KxN) from TRANSPOSED (N x K, ldT) LDS tile at `baseT`
// (baseT already offset to tile row nt*16, col kt*32).
// lane: n = lane&15, kb = (lane>=16)?8:0; elems 0..7 = K kb..kb+7, 8..15 = K 16+kb..23+kb.
__device__ __forceinline__ v16bf load_b_tr(const __bf16* baseT, int ldT) {
    const int lane = threadIdx.x & 31;
    const __bf16* p = baseT + (lane & 15) * ldT + ((lane >> 4) << 3);
    frag16 u;
    u.q[0] = *(const uint4*)(p);
    u.q[1] = *(const uint4*)(p + 16);
    return u.f;
}

__global__ __launch_bounds__(256)
void molgan_disc_kernel(const float* __restrict__ adj,  const float* __restrict__ node,
                        const float* __restrict__ W1,   const float* __restrict__ b1,
                        const float* __restrict__ W2,   const float* __restrict__ b2,
                        const float* __restrict__ Wsw,  const float* __restrict__ bsw,
                        const float* __restrict__ Wtw,  const float* __restrict__ btw,
                        const float* __restrict__ Wl1,  const float* __restrict__ bl1,
                        const float* __restrict__ Wl2,  const float* __restrict__ bl2,
                        const float* __restrict__ Wo,   const float* __restrict__ bo,
                        float* __restrict__ out, int Bcount) {
    __shared__ __align__(16) __bf16 sA [NN * STR];   // reduced adjacency, row-major (A side)
    __shared__ __align__(16) __bf16 sXT[NN * STR];   // transposed activations (B side): h1T, then h2T
    __shared__ __align__(16) __bf16 sY [NN * STR];   // row-major activations (A side): h, then ann2
    __shared__ __align__(16) __bf16 sWs[NN * WSTR];  // W2T (flat, early) then WsT
    __shared__ __align__(16) __bf16 sWt[NN * WSTR];  // WtT
    __shared__ float gp[128];
    __shared__ float sg[128];
    __shared__ float sg2[128];

    const int b    = blockIdx.x;
    const int tid  = threadIdx.x;
    const int wave = tid >> 5;
    const int lane = tid & 31;
    const int n_in = lane & 15;
    const int mloc = (lane >> 4) << 3;               // 0 or 8 (C/D row sub-base)

    const float* adjb  = adj  + (size_t)b * NN * NN * 5;
    const float* nodeb = node + (size_t)b * NN * 5;

    if (tid < 128) gp[tid] = 0.0f;

    // ---- A[i][j] = sum_{d=1..4} adj[b,i,j,d]; 4 elems/iter via 5 aligned float4 ----
    for (int f4 = tid; f4 < (NN * NN) / 4; f4 += 256) {
        const float4* q = (const float4*)(adjb + (size_t)f4 * 20);
        const float4 q0 = q[0], q1 = q[1], q2 = q[2], q3 = q[3], q4 = q[4];
        pack4 pk;
        pk.h[0] = f2bf16(q0.y + q0.z + q0.w + q1.x);
        pk.h[1] = f2bf16(q1.z + q1.w + q2.x + q2.y);
        pk.h[2] = f2bf16(q2.w + q3.x + q3.y + q3.z);
        pk.h[3] = f2bf16(q4.x + q4.y + q4.z + q4.w);
        const int e0 = f4 * 4;
        *(uint2*)&sA[(e0 >> 7) * STR + (e0 & 127)] = pk.q;
    }

    // ---- h1T[c][i] = (node @ W1 + b1)[i][c]   (B-side: stored transposed) ----
    for (int f = tid; f < NN * 128; f += 256) {
        const int i = f & 127, c = f >> 7;
        float acc = b1[c];
#pragma unroll
        for (int m = 0; m < 5; ++m) acc += nodeb[i * 5 + m] * W1[m * 128 + c];
        sXT[c * STR + i] = f2bf16(acc);
    }

    // ---- stage W2T[n][k] = W2[k][n] into sWs (flat, stride 128) ----
    for (int f = tid; f < 64 * 128; f += 256) {
        const int n = f & 63, k = f >> 6;
        sWs[n * 128 + k] = f2bf16(W2[k * 64 + n]);
    }
    __syncthreads();

    // ---- h = A @ h1 + h1  (128x128x128):  sA x sXT -> sY (row-major) ----
    {
        const int mt = wave;
        v16bf afr[4];
#pragma unroll
        for (int kt = 0; kt < 4; ++kt)
            afr[kt] = load_a_rm(&sA[(mt * 16) * STR + kt * 32], STR);
        const int mbase = mt * 16 + mloc;
        for (int nt = 0; nt < 8; ++nt) {
            v8f c = {};
#pragma unroll
            for (int kt = 0; kt < 4; ++kt)
                c = wmma_bf16(afr[kt], load_b_tr(&sXT[(nt * 16) * STR + kt * 32], STR), c);
            const int n = nt * 16 + n_in;
            pack8 res; res.q = *(const uint4*)&sXT[n * STR + mbase];  // h1 residual (contig)
#pragma unroll
            for (int v = 0; v < 8; ++v)
                sY[(mbase + v) * STR + n] = f2bf16(c[v] + bf2f(res.h[v]));
        }
    }
    __syncthreads();

    // ---- h2T = (h @ W2 + b2)^T  (128x128x64):  sY x W2T -> sXT (packed 16B stores) ----
    {
        const int mt = wave;
        v16bf afr[4];
#pragma unroll
        for (int kt = 0; kt < 4; ++kt)
            afr[kt] = load_a_rm(&sY[(mt * 16) * STR + kt * 32], STR);
        const int mbase = mt * 16 + mloc;
        for (int nt = 0; nt < 4; ++nt) {
            v8f c = {};
#pragma unroll
            for (int kt = 0; kt < 4; ++kt)
                c = wmma_bf16(afr[kt], load_b_tr(&sWs[(nt * 16) * 128 + kt * 32], 128), c);
            const int n = nt * 16 + n_in;
            const float bias = b2[n];
            pack8 pk;
#pragma unroll
            for (int v = 0; v < 8; ++v) pk.h[v] = f2bf16(c[v] + bias);
            *(uint4*)&sXT[n * STR + mbase] = pk.q;                    // transposed store
        }
    }
    __syncthreads();

    // ---- h3 = A @ h2 + h2 -> sY[:,0:64];  ann2 = [h3 | node | 0..] (cols 64..95) ----
    {
        const int mt = wave;
        v16bf afr[4];
#pragma unroll
        for (int kt = 0; kt < 4; ++kt)
            afr[kt] = load_a_rm(&sA[(mt * 16) * STR + kt * 32], STR);
        const int mbase = mt * 16 + mloc;
        for (int nt = 0; nt < 4; ++nt) {
            v8f c = {};
#pragma unroll
            for (int kt = 0; kt < 4; ++kt)
                c = wmma_bf16(afr[kt], load_b_tr(&sXT[(nt * 16) * STR + kt * 32], STR), c);
            const int n = nt * 16 + n_in;
            pack8 res; res.q = *(const uint4*)&sXT[n * STR + mbase];  // h2 residual
#pragma unroll
            for (int v = 0; v < 8; ++v)
                sY[(mbase + v) * STR + n] = f2bf16(c[v] + bf2f(res.h[v]));
        }
    }
    // ann2 pad: cols 64..68 = node, 69..95 = 0 (K padded to 96 for the pooled GEMM)
    for (int f = tid; f < NN * 32; f += 256) {
        const int i = f >> 5, c = 64 + (f & 31);
        sY[i * STR + c] = f2bf16((c < 69) ? nodeb[i * 5 + (c - 64)] : 0.0f);
    }
    // stage WsT / WtT (N x K, K padded 69->96, zero-filled), coalesced global reads
    for (int f = tid; f < NN * WSTR; f += 256) {
        const int n = f & 127, k = f >> 7;
        sWs[n * WSTR + k] = f2bf16((k < 69) ? Wsw[k * 128 + n] : 0.0f);
        sWt[n * WSTR + k] = f2bf16((k < 69) ? Wtw[k * 128 + n] : 0.0f);
    }
    __syncthreads();

    // ---- gated pool: gp[c] = sum_i sigmoid(ann2@Ws+bs)*tanh(ann2@Wt+bt), K=96 ----
    {
        const int mt = wave;
        v16bf afr[3];
#pragma unroll
        for (int kt = 0; kt < 3; ++kt)
            afr[kt] = load_a_rm(&sY[(mt * 16) * STR + kt * 32], STR);
        for (int nt = 0; nt < 8; ++nt) {
            v8f cs = {}, ct = {};
#pragma unroll
            for (int kt = 0; kt < 3; ++kt) {
                cs = wmma_bf16(afr[kt], load_b_tr(&sWs[(nt * 16) * WSTR + kt * 32], WSTR), cs);
                ct = wmma_bf16(afr[kt], load_b_tr(&sWt[(nt * 16) * WSTR + kt * 32], WSTR), ct);
            }
            const int n = nt * 16 + n_in;
            const float bias_s = bsw[n];
            const float bias_t = btw[n];
            float part = 0.0f;
#pragma unroll
            for (int v = 0; v < 8; ++v) {
                const float sv = 1.0f / (1.0f + __expf(-(cs[v] + bias_s)));
                const float tv = tanhf(ct[v] + bias_t);
                part += sv * tv;
            }
            atomicAdd(&gp[n], part);   // ds_add_f32
        }
    }
    __syncthreads();

    // ---- head: g = tanh(gp); g2 = g@Wl1+bl1; g3 = g2@Wl2+bl2; out = g3@Wo+bo ----
    if (tid < 128) sg[tid] = tanhf(gp[tid]);
    __syncthreads();
    if (tid < 128) {
        float acc = bl1[tid];
#pragma unroll 4
        for (int k = 0; k < 128; ++k) acc += sg[k] * Wl1[k * 128 + tid];
        sg2[tid] = acc;
    }
    __syncthreads();
    if (tid < 64) {
        float acc = bl2[tid];
#pragma unroll 4
        for (int k = 0; k < 128; ++k) acc += sg2[k] * Wl2[k * 64 + tid];
        out[Bcount + b * 64 + tid] = acc;   // returned g [B,64]
        gp[tid] = acc;                      // stash g3
    }
    __syncthreads();
    if (tid == 0) {
        float acc = bo[0];
#pragma unroll 4
        for (int k = 0; k < 64; ++k) acc += gp[k] * Wo[k];
        out[b] = acc;                       // returned out [B,1]
    }
}

extern "C" void kernel_launch(void* const* d_in, const int* in_sizes, int n_in,
                              void* d_out, int out_size, void* d_ws, size_t ws_size,
                              hipStream_t stream) {
    (void)n_in; (void)out_size; (void)d_ws; (void)ws_size;
    const float* adj  = (const float*)d_in[0];
    const float* node = (const float*)d_in[1];
    const float* W1   = (const float*)d_in[2];
    const float* b1   = (const float*)d_in[3];
    const float* W2   = (const float*)d_in[4];
    const float* b2   = (const float*)d_in[5];
    const float* Ws   = (const float*)d_in[6];
    const float* bs   = (const float*)d_in[7];
    const float* Wt   = (const float*)d_in[8];
    const float* bt   = (const float*)d_in[9];
    const float* Wl1  = (const float*)d_in[10];
    const float* bl1  = (const float*)d_in[11];
    const float* Wl2  = (const float*)d_in[12];
    const float* bl2  = (const float*)d_in[13];
    const float* Wo   = (const float*)d_in[14];
    const float* bo   = (const float*)d_in[15];

    const int Bcount = in_sizes[0] / (NN * NN * 5);   // 512

    molgan_disc_kernel<<<dim3(Bcount), dim3(256), 0, stream>>>(
        adj, node, W1, b1, W2, b2, Ws, bs, Wt, bt,
        Wl1, bl1, Wl2, bl2, Wo, bo, (float*)d_out, Bcount);
}
